// ARC_56461640073584
// MI455X (gfx1250) — matrix operand order repaired and checked by reading, not compiled, not run on hardware
//
#include <hip/hip_runtime.h>
#include <hip/hip_bf16.h>

// ---------------------------------------------------------------------------
// Shapes
#define BB   2
#define TT   2304          // 128 + 2048 + 128
#define DIN  1024
#define HH   8
#define EE   64
#define SEG1 128
#define SEG2 2176

typedef __attribute__((ext_vector_type(16))) __bf16 v16bf;
typedef __attribute__((ext_vector_type(8)))  float  v8f;

union BF16Frag { uint4 q[2]; v16bf v; };

__device__ __forceinline__ unsigned short f2bf(float f) {
    unsigned int u = __float_as_uint(f);
    unsigned int r = u + 0x7FFFu + ((u >> 16) & 1u);   // round-to-nearest-even
    return (unsigned short)(r >> 16);
}

// A-matrix 16x32 bf16 fragment: lane m=l%16, g=l/16.
// halves 0..7  = K in [g*16,   g*16+8)
// halves 8..15 = K in [g*16+8, g*16+16)
__device__ __forceinline__ v16bf load_a(const unsigned short* rowk, int g) {
    BF16Frag u;
    u.q[0] = *(const uint4*)(rowk + g * 16);
    u.q[1] = *(const uint4*)(rowk + g * 16 + 8);
    return u.v;
}
// B-matrix 32x16 bf16 fragment: lane n=l%16, kg=l/16; 16 contiguous K values.
__device__ __forceinline__ v16bf load_b(const unsigned short* p) {
    BF16Frag u;
    u.q[0] = *(const uint4*)(p);
    u.q[1] = *(const uint4*)(p + 8);
    return u.v;
}
__device__ __forceinline__ v8f wmma_bf16(v16bf a, v16bf b, v8f c) {
    return __builtin_amdgcn_wmma_f32_16x16x32_bf16(false, a, false, b, (short)0, c,
                                                   false, false);
}
__device__ __forceinline__ v8f vzero8() {
    v8f z;
#pragma unroll
    for (int i = 0; i < 8; ++i) z[i] = 0.0f;
    return z;
}

// ---------------------------------------------------------------------------
// K0: LayerNorm each (b,t) row of x over Din, store bf16 normalized x.
__global__ __launch_bounds__(256) void k_layernorm(const float* __restrict__ x,
                                                   unsigned short* __restrict__ nrm) {
    const int row = blockIdx.x;                     // b*T + t
    const int tid = threadIdx.x;
    const float* xr = x + (size_t)row * DIN;
    float4 xv = *(const float4*)(xr + tid * 4);

    __shared__ float s1[256], s2[256];
    float s  = xv.x + xv.y + xv.z + xv.w;
    float sq = xv.x * xv.x + xv.y * xv.y + xv.z * xv.z + xv.w * xv.w;
    s1[tid] = s; s2[tid] = sq;
    __syncthreads();
#pragma unroll
    for (int o = 128; o > 0; o >>= 1) {
        if (tid < o) { s1[tid] += s1[tid + o]; s2[tid] += s2[tid + o]; }
        __syncthreads();
    }
    const float mean = s1[0] * (1.0f / DIN);
    const float var  = s2[0] * (1.0f / DIN) - mean * mean;
    const float rs   = rsqrtf(var + 1e-5f);

    unsigned short* o = nrm + (size_t)row * DIN + tid * 4;
    o[0] = f2bf((xv.x - mean) * rs);
    o[1] = f2bf((xv.y - mean) * rs);
    o[2] = f2bf((xv.z - mean) * rs);
    o[3] = f2bf((xv.w - mean) * rs);
}

// ---------------------------------------------------------------------------
// K1: fold per-head affine into the weights.
//   Wt[h][e][d] = bf16(g[h][d] * W[h][d][e])       (e-major for B-frag loads)
//   bias[h][e]  = sum_d b[h][d] * W[h][d][e]
__global__ __launch_bounds__(256) void k_fuse(const float* __restrict__ g,
                                              const float* __restrict__ bt,
                                              const float* __restrict__ W,
                                              unsigned short* __restrict__ Wt,
                                              float* __restrict__ bias) {
    const int h = blockIdx.x >> 6;
    const int e = blockIdx.x & 63;
    const int tid = threadIdx.x;
    const float* gh = g  + (size_t)h * DIN;
    const float* bh = bt + (size_t)h * DIN;
    const float* Wh = W  + (size_t)h * DIN * EE + e;     // stride EE over d
    unsigned short* out = Wt + ((size_t)h * EE + e) * DIN;

    float partial = 0.0f;
#pragma unroll
    for (int i = 0; i < 4; ++i) {
        const int d = tid * 4 + i;
        const float w = Wh[(size_t)d * EE];
        out[d] = f2bf(gh[d] * w);
        partial += bh[d] * w;
    }
    __shared__ float sb[256];
    sb[tid] = partial;
    __syncthreads();
#pragma unroll
    for (int o = 128; o > 0; o >>= 1) {
        if (tid < o) sb[tid] += sb[tid + o];
        __syncthreads();
    }
    if (tid == 0) bias[h * EE + e] = sb[0];
}

// ---------------------------------------------------------------------------
// K2: projection GEMM. Block = 128 thr (4 waves), 64 rows x 64 cols per block.
// K-loop fully unrolled: every fragment is a distinct SSA value, so there are
// no loop-carried registers -> no v_mov copies, and the scheduler can hoist
// loads many WMMAs ahead (deep s_wait_loadcnt pipelining).
__global__ __launch_bounds__(128) void k_proj(const unsigned short* __restrict__ nrm,
                                              const unsigned short* __restrict__ Wt,
                                              const float* __restrict__ bias,
                                              unsigned short* __restrict__ qO,
                                              unsigned short* __restrict__ kO,
                                              unsigned short* __restrict__ vT) {
    const int tile = blockIdx.x, p = blockIdx.y, bh = blockIdx.z;
    const int b = bh >> 3, h = bh & 7;
    const int t0 = tile * 64;
    const int seg = (t0 < SEG1) ? 0 : (t0 < SEG2 ? 1 : 2);
    const int j = ((seg == 0) ? 3 : (seg == 1) ? 0 : 6) + p;

    const int lane = threadIdx.x & 31, w = threadIdx.x >> 5;
    const int n = lane & 15, g = lane >> 4;             // A: m=n, g ; B: n, kg=g

    const unsigned short* arow = nrm + ((size_t)b * TT + (t0 + w * 16 + n)) * DIN;
    const unsigned short* wcol = Wt + ((size_t)(j * HH + h) * EE + n) * DIN + g * 16;
    const float* bs = bias + (j * HH + h) * EE;

    v8f acc[4];
#pragma unroll
    for (int nt = 0; nt < 4; ++nt) acc[nt] = vzero8();

#pragma unroll
    for (int kk = 0; kk < DIN; kk += 32) {
        const v16bf a = load_a(arow + kk, g);
#pragma unroll
        for (int nt = 0; nt < 4; ++nt) {
            const v16bf bf = load_b(wcol + nt * 16 * DIN + kk);
            acc[nt] = wmma_bf16(a, bf, acc[nt]);
        }
    }

    const float qscale = (p == 0) ? 0.125f : 1.0f;      // fold 1/sqrt(64) into q
#pragma unroll
    for (int nt = 0; nt < 4; ++nt) {
        const float bv = bs[nt * 16 + n];
#pragma unroll
        for (int r = 0; r < 8; ++r) {
            const int tt = t0 + w * 16 + r + 8 * g;     // C layout: M = r + 8*(l/16)
            const int e  = nt * 16 + n;                 //           N = l%16
            const float val = (acc[nt][r] + bv) * qscale;
            if (p == 0)
                qO[((size_t)bh * TT + tt) * EE + e] = f2bf(val);
            else if (p == 1)
                kO[((size_t)bh * TT + tt) * EE + e] = f2bf(val);
            else
                vT[((size_t)bh * EE + e) * TT + tt] = f2bf(val);
        }
    }
}

// ---------------------------------------------------------------------------
// K3: attention with online softmax. Block = 128 thr (4 waves), 64 q-rows.
// scores = q·kT (scale folded) + 1.0 on lower triangle; softmax over all T.
// Fragment rotation is inside fully-unrolled loops (no copy overhead) and the
// next j-tile of K/V is prefetched (global_prefetch_b8) over the softmax VALU.
__global__ __launch_bounds__(128) void k_attn(const unsigned short* __restrict__ qI,
                                              const unsigned short* __restrict__ kI,
                                              const unsigned short* __restrict__ vT,
                                              float* __restrict__ out) {
    const int tile = blockIdx.x, bh = blockIdx.y;
    const int b = bh >> 3, h = bh & 7;
    const int t0 = tile * 64;
    const int lane = threadIdx.x & 31, w = threadIdx.x >> 5;
    const int n = lane & 15, g = lane >> 4;

    __shared__ __align__(16) unsigned short ldsP[4][16][72];

    // Q fragments (A layout), 2 K-chunks of 32 over head dim 64
    const unsigned short* qrow = qI + ((size_t)bh * TT + (t0 + w * 16 + n)) * EE;
    v16bf qf[2];
#pragma unroll
    for (int s = 0; s < 2; ++s) qf[s] = load_a(qrow + s * 32, g);

    float mstat[8], lstat[8];
    v8f o[4];
#pragma unroll
    for (int r = 0; r < 8; ++r) { mstat[r] = -1e30f; lstat[r] = 0.0f; }
#pragma unroll
    for (int nt = 0; nt < 4; ++nt) o[nt] = vzero8();

    for (int j0 = 0; j0 < TT; j0 += 64) {
        // ---- prefetch next j-tile of K and V into cache (global_prefetch)
        const int j0n = (j0 + 64 < TT) ? (j0 + 64) : 0;
        __builtin_prefetch(kI + ((size_t)bh * TT + j0n + lane) * EE, 0, 0);
        __builtin_prefetch(kI + ((size_t)bh * TT + j0n + 32 + lane) * EE, 0, 0);
        __builtin_prefetch(vT + ((size_t)bh * EE + lane) * TT + j0n, 0, 0);
        __builtin_prefetch(vT + ((size_t)bh * EE + 32 + lane) * TT + j0n, 0, 0);

        // ---- scores tile S (16 q-rows x 64 keys), K-frag rotation --------
        const unsigned short* kb0 =
            kI + ((size_t)bh * TT + j0 + n) * EE + g * 16;
        v16bf k_c0 = load_b(kb0);
        v16bf k_c1 = load_b(kb0 + 32);
        v8f S[4];
#pragma unroll
        for (int jt = 0; jt < 4; ++jt) {
            const int jn = (jt < 3) ? (jt + 1) * 16 * EE : 0;   // next keys (wrap)
            v16bf k_n0 = load_b(kb0 + jn);
            v16bf k_n1 = load_b(kb0 + jn + 32);
            v8f c = vzero8();
            c = wmma_bf16(qf[0], k_c0, c);
            c = wmma_bf16(qf[1], k_c1, c);
            const int tk = j0 + jt * 16 + n;
#pragma unroll
            for (int r = 0; r < 8; ++r) {
                const int tq = t0 + w * 16 + r + 8 * g;
                if (tk <= tq) c[r] += 1.0f;             // tril(ones) is ADDED
            }
            S[jt] = c;
            k_c0 = k_n0; k_c1 = k_n1;
        }
        // ---- online softmax ---------------------------------------------
#pragma unroll
        for (int r = 0; r < 8; ++r) {
            float mx = S[0][r];
#pragma unroll
            for (int jt = 1; jt < 4; ++jt) mx = fmaxf(mx, S[jt][r]);
#pragma unroll
            for (int d = 1; d < 16; d <<= 1)
                mx = fmaxf(mx, __shfl_xor(mx, d, 32));
            const float nm = fmaxf(mstat[r], mx);
            const float sc = __expf(mstat[r] - nm);
            mstat[r] = nm;
            lstat[r] *= sc;
#pragma unroll
            for (int nt = 0; nt < 4; ++nt) o[nt][r] *= sc;
            float rsum = 0.0f;
#pragma unroll
            for (int jt = 0; jt < 4; ++jt) {
                const float pv = __expf(S[jt][r] - nm);
                S[jt][r] = pv;
                rsum += pv;
            }
#pragma unroll
            for (int d = 1; d < 16; d <<= 1)
                rsum += __shfl_xor(rsum, d, 32);
            lstat[r] += rsum;
        }
        // ---- P to LDS (C layout -> A layout transpose) -------------------
#pragma unroll
        for (int jt = 0; jt < 4; ++jt)
#pragma unroll
            for (int r = 0; r < 8; ++r)
                ldsP[w][r + 8 * g][jt * 16 + n] = f2bf(S[jt][r]);
        __syncthreads();
        // ---- O += P · V  (V-frag rotation) -------------------------------
        v16bf pa = load_a(&ldsP[w][n][0], g);
#pragma unroll
        for (int s = 0; s < 2; ++s) {
            v16bf pa_n = load_a(&ldsP[w][n][s ? 0 : 32], g);
            const unsigned short* vb0 =
                vT + ((size_t)bh * EE + n) * TT + j0 + s * 32 + g * 16;
            v16bf vb_c = load_b(vb0);
#pragma unroll
            for (int nt = 0; nt < 4; ++nt) {
                v16bf vb_n = load_b(vb0 + ((nt < 3) ? (nt + 1) * 16 * TT : 0));
                o[nt] = wmma_bf16(pa, vb_c, o[nt]);
                vb_c = vb_n;
            }
            pa = pa_n;
        }
        __syncthreads();
    }
    // ---- normalize + store out[b][t][h*64+e] -----------------------------
#pragma unroll
    for (int nt = 0; nt < 4; ++nt)
#pragma unroll
        for (int r = 0; r < 8; ++r) {
            const int tt = t0 + w * 16 + r + 8 * g;
            const int e  = nt * 16 + n;
            out[((size_t)b * TT + tt) * (HH * EE) + h * EE + e] =
                o[nt][r] / lstat[r];
        }
}

// ---------------------------------------------------------------------------
extern "C" void kernel_launch(void* const* d_in, const int* in_sizes, int n_in,
                              void* d_out, int out_size, void* d_ws, size_t ws_size,
                              hipStream_t stream) {
    const float* x = (const float*)d_in[0];
    // ln params: 1:g 2:b (mid) 3:gs 4:bs (start) 5:ge 6:be (end)
    // weights 7..15: Wq,Wk,Wv, Wq_s,Wk_s,Wv_s, Wq_e,Wk_e,Wv_e
    const float* lnG[9] = {(const float*)d_in[1], (const float*)d_in[1], (const float*)d_in[1],
                           (const float*)d_in[3], (const float*)d_in[3], (const float*)d_in[3],
                           (const float*)d_in[5], (const float*)d_in[5], (const float*)d_in[5]};
    const float* lnB[9] = {(const float*)d_in[2], (const float*)d_in[2], (const float*)d_in[2],
                           (const float*)d_in[4], (const float*)d_in[4], (const float*)d_in[4],
                           (const float*)d_in[6], (const float*)d_in[6], (const float*)d_in[6]};
    float* out = (float*)d_out;

    // workspace layout (bytes)
    const size_t sz_norm = (size_t)BB * TT * DIN * 2;          //  9,437,184
    const size_t sz_wt   = (size_t)9 * HH * EE * DIN * 2;      //  9,437,184
    const size_t sz_bias = (size_t)9 * HH * EE * 4;            //     18,432
    const size_t sz_qkv  = (size_t)BB * HH * TT * EE * 2;      //  4,718,592
    char* ws = (char*)d_ws;
    unsigned short* norm = (unsigned short*)(ws);
    unsigned short* Wt   = (unsigned short*)(ws + sz_norm);
    float*          bias = (float*)(ws + sz_norm + sz_wt);
    unsigned short* qB   = (unsigned short*)(ws + sz_norm + sz_wt + sz_bias);
    unsigned short* kB   = (unsigned short*)((char*)qB + sz_qkv);
    unsigned short* vB   = (unsigned short*)((char*)kB + sz_qkv);
    if (ws_size < sz_norm + sz_wt + sz_bias + 3 * sz_qkv) return;

    k_layernorm<<<BB * TT, 256, 0, stream>>>(x, norm);

    for (int j = 0; j < 9; ++j) {
        const float* W = (const float*)d_in[7 + j];
        k_fuse<<<HH * EE, 256, 0, stream>>>(lnG[j], lnB[j], W,
                                            Wt + (size_t)j * HH * EE * DIN,
                                            bias + (size_t)j * HH * EE);
    }

    k_proj<<<dim3(TT / 64, 3, BB * HH), 128, 0, stream>>>(norm, Wt, bias, qB, kB, vB);
    k_attn<<<dim3(TT / 64, BB * HH), 128, 0, stream>>>(qB, kB, vB, out);
}
